// HyperbolicLinearPoincare_18743237279921
// MI455X (gfx1250) — matrix operand order repaired and checked by reading, not compiled, not run on hardware
//
#include <hip/hip_runtime.h>
#include <cstdint>

typedef __attribute__((ext_vector_type(16))) __bf16 v16bf;
typedef __attribute__((ext_vector_type(8)))  float  v8f;

#define IN_DIM       128
#define OUT_DIM      128
#define ROWS_PER_BLK 128
#define WSTR 136   // bf16 elems per weight row in LDS (272B, 16B-padded)
#define VSTR 136   // bf16 elems per v row in LDS
#define OSTR 132   // f32 elems per out row in LDS (528B, 16B-aligned)

#define SMEM_W_BYTES    (OUT_DIM * WSTR * 2)            // 34816
#define SMEM_BIAS_BYTES (OUT_DIM * 4)                   // 512
#define SMEM_BUF_BYTES  (ROWS_PER_BLK * OSTR * 4)       // 67584 (aliases v-rows)
#define SMEM_TOTAL (SMEM_W_BYTES + SMEM_BIAS_BYTES + SMEM_BUF_BYTES)

__device__ __forceinline__ unsigned short f2bf(float f) {
  unsigned int u = __float_as_uint(f);
  u += 0x7FFFu + ((u >> 16) & 1u);   // round-to-nearest-even
  return (unsigned short)(u >> 16);
}

struct Bits256 { unsigned int u[8]; };

// A-matrix fragment (16x32 bf16): lane L = row L%16; VGPR0-3 hold
// K = (L/16)*8 + 0..7, VGPR4-7 hold K = 16 + (L/16)*8 + 0..7
// -> two NON-contiguous b128 LDS loads (p and p+16 halfs).
__device__ __forceinline__ v16bf load_frag_a(const unsigned short* p) {
  uint4 a = *reinterpret_cast<const uint4*>(p);
  uint4 b = *reinterpret_cast<const uint4*>(p + 16);
  Bits256 t;
  t.u[0] = a.x; t.u[1] = a.y; t.u[2] = a.z; t.u[3] = a.w;
  t.u[4] = b.x; t.u[5] = b.y; t.u[6] = b.z; t.u[7] = b.w;
  return __builtin_bit_cast(v16bf, t);
}

// B-matrix fragment (32x16 bf16): lane L = col L%16; VGPR0-3 hold
// K = (L/16)*8 + 0..7, VGPR4-7 hold K = 16 + (L/16)*8 + 0..7
// -> two NON-contiguous b128 LDS loads (p and p+16 halfs).
__device__ __forceinline__ v16bf load_frag_b(const unsigned short* p) {
  uint4 a = *reinterpret_cast<const uint4*>(p);
  uint4 b = *reinterpret_cast<const uint4*>(p + 16);
  Bits256 t;
  t.u[0] = a.x; t.u[1] = a.y; t.u[2] = a.z; t.u[3] = a.w;
  t.u[4] = b.x; t.u[5] = b.y; t.u[6] = b.z; t.u[7] = b.w;
  return __builtin_bit_cast(v16bf, t);
}

__global__ __launch_bounds__(256)
void hyp_linear_poincare_kernel(const float* __restrict__ x,
                                const float* __restrict__ weight,
                                const float* __restrict__ bias,
                                float* __restrict__ out,
                                int nrows)
{
  extern __shared__ __align__(16) unsigned char smem[];
  unsigned short* w_lds   = reinterpret_cast<unsigned short*>(smem);
  float*          bias_ld = reinterpret_cast<float*>(smem + SMEM_W_BYTES);
  unsigned char*  buf     = smem + SMEM_W_BYTES + SMEM_BIAS_BYTES;
  unsigned short* v_lds   = reinterpret_cast<unsigned short*>(buf);  // phase 1-2a
  float*          out_lds = reinterpret_cast<float*>(buf);           // phase 2b-3 (aliased)

  const int   tid  = threadIdx.x;
  const float MAXN = 1.0f - 1e-5f;

  // ---- Phase 0: weight f32 -> bf16 into LDS (row-major OUTxIN, padded) ----
  for (int idx = tid; idx < OUT_DIM * IN_DIM; idx += 256) {
    int o = idx >> 7, i = idx & 127;
    w_lds[o * WSTR + i] = f2bf(weight[idx]);
  }
  if (tid < OUT_DIM) bias_ld[tid] = bias[tid];

  // ---- Phase 1: logmap_0, half-row per thread ----
  const int row = tid >> 1;       // 0..127 within tile
  const int hf  = tid & 1;        // which 64-elem half
  long long grow  = (long long)blockIdx.x * ROWS_PER_BLK + row;
  bool      valid = grow < (long long)nrows;
  long long gbase = valid ? grow : 0;

  const float4* xp4 = reinterpret_cast<const float4*>(x + gbase * IN_DIM + hf * 64);
  float4 xv[16];
  float ss = 0.f;
  #pragma unroll
  for (int i = 0; i < 16; ++i) {
    float4 v = xp4[i];
    xv[i] = v;
    ss += v.x*v.x + v.y*v.y + v.z*v.z + v.w*v.w;
  }
  ss += __shfl_xor(ss, 1);                       // partner holds other half of row
  float n1  = fmaxf(sqrtf(ss), 1e-15f);
  float arg = fminf(n1, MAXN);
  float at  = 0.5f * logf((1.f + arg) / (1.f - arg));   // atanh
  float s1  = at / n1;

  unsigned short* vrow = v_lds + row * VSTR + hf * 64;
  #pragma unroll
  for (int i = 0; i < 16; ++i) {
    float4 v = xv[i];
    unsigned int lo = (unsigned int)f2bf(v.x * s1) | ((unsigned int)f2bf(v.y * s1) << 16);
    unsigned int hi = (unsigned int)f2bf(v.z * s1) | ((unsigned int)f2bf(v.w * s1) << 16);
    *reinterpret_cast<uint2*>(vrow + i * 4) = make_uint2(lo, hi);
  }
  __syncthreads();

  // ---- Phase 2: GEMM  out = v @ W^T  via bf16 WMMA ----
  const int wave  = tid >> 5;
  const int lane  = tid & 31;
  const int lrow  = lane & 15;    // A-row / B-col / D-col within 16-tile
  const int lhalf = lane >> 4;    // K-half selector
  const int wrow  = wave * 16;    // this wave's row stripe base

  v16bf afrag[4];
  #pragma unroll
  for (int kt = 0; kt < 4; ++kt)
    afrag[kt] = load_frag_a(v_lds + (wrow + lrow) * VSTR + kt * 32 + lhalf * 8);

  __syncthreads();  // A frags now in regs; v_lds dead -> out_lds aliasing is safe

  #pragma unroll
  for (int nt = 0; nt < 8; ++nt) {
    v8f acc = {0.f, 0.f, 0.f, 0.f, 0.f, 0.f, 0.f, 0.f};
    #pragma unroll
    for (int kt = 0; kt < 4; ++kt) {
      // B[k][n] = W[n][k]: row n of the LDS weight tile; lane L needs
      // K = kt*32 + (L/16)*8 + 0..7 and kt*32 + 16 + (L/16)*8 + 0..7
      v16bf bfrag = load_frag_b(w_lds + (nt * 16 + lrow) * WSTR + kt * 32 + lhalf * 8);
      acc = __builtin_amdgcn_wmma_f32_16x16x32_bf16(
                false, afrag[kt], false, bfrag, (short)0, acc, false, false);
    }
    // D layout: lane L -> N = L%16 ; VGPR j -> M = j + 8*(L/16)
    #pragma unroll
    for (int j = 0; j < 8; ++j)
      out_lds[(wrow + j + 8 * lhalf) * OSTR + nt * 16 + lrow] = acc[j];
  }
  __syncthreads();

  // ---- Phase 3: expmap_0 + project + mobius_add(bias) + project ----
  const float4* orow = reinterpret_cast<const float4*>(out_lds + row * OSTR + hf * 64);
  float4 mv[16];
  float ss2 = 0.f;
  #pragma unroll
  for (int i = 0; i < 16; ++i) {
    float4 v = orow[i];
    mv[i] = v;
    ss2 += v.x*v.x + v.y*v.y + v.z*v.z + v.w*v.w;
  }
  ss2 += __shfl_xor(ss2, 1);
  float n2 = fmaxf(sqrtf(ss2), 1e-15f);
  float th = tanhf(n2);
  float es = th / n2;                              // expmap scale
  float ps = (th > MAXN) ? (MAXN / th) : 1.0f;     // projection of expmap result
  float sc = es * ps;

  const float4* brow = reinterpret_cast<const float4*>(bias_ld + hf * 64);
  float x2 = 0.f, xy = 0.f, y2 = 0.f;
  #pragma unroll
  for (int i = 0; i < 16; ++i) {
    float4 m = mv[i];
    m.x *= sc; m.y *= sc; m.z *= sc; m.w *= sc;
    mv[i] = m;
    float4 b = brow[i];
    x2 += m.x*m.x + m.y*m.y + m.z*m.z + m.w*m.w;
    xy += m.x*b.x + m.y*b.y + m.z*b.z + m.w*b.w;
    y2 += b.x*b.x + b.y*b.y + b.z*b.z + b.w*b.w;
  }
  x2 += __shfl_xor(x2, 1);
  xy += __shfl_xor(xy, 1);
  y2 += __shfl_xor(y2, 1);

  float c1  = 1.f + 2.f * xy + y2;                 // coeff on m   (C=1)
  float c2  = 1.f - x2;                            // coeff on bias
  float den = fmaxf(1.f + 2.f * xy + x2 * y2, 1e-15f);
  float inv = 1.f / den;

  float rr = 0.f;
  #pragma unroll
  for (int i = 0; i < 16; ++i) {
    float4 b = brow[i];
    float4 r;
    r.x = (c1 * mv[i].x + c2 * b.x) * inv;
    r.y = (c1 * mv[i].y + c2 * b.y) * inv;
    r.z = (c1 * mv[i].z + c2 * b.z) * inv;
    r.w = (c1 * mv[i].w + c2 * b.w) * inv;
    mv[i] = r;
    rr += r.x*r.x + r.y*r.y + r.z*r.z + r.w*r.w;
  }
  rr += __shfl_xor(rr, 1);
  float rn = fmaxf(sqrtf(rr), 1e-15f);
  float fs = (rn > MAXN) ? (MAXN / rn) : 1.0f;     // final projection

  if (valid) {
    float4* op = reinterpret_cast<float4*>(out + grow * OUT_DIM + hf * 64);
    #pragma unroll
    for (int i = 0; i < 16; ++i) {
      float4 r = mv[i];
      r.x *= fs; r.y *= fs; r.z *= fs; r.w *= fs;
      op[i] = r;
    }
  }
}

extern "C" void kernel_launch(void* const* d_in, const int* in_sizes, int n_in,
                              void* d_out, int out_size, void* d_ws, size_t ws_size,
                              hipStream_t stream) {
  (void)n_in; (void)out_size; (void)d_ws; (void)ws_size;
  const float* x      = (const float*)d_in[0];
  const float* weight = (const float*)d_in[1];
  const float* bias   = (const float*)d_in[2];
  float*       out    = (float*)d_out;

  int nrows   = in_sizes[0] / IN_DIM;
  int nblocks = (nrows + ROWS_PER_BLK - 1) / ROWS_PER_BLK;

  hipLaunchKernelGGL(hyp_linear_poincare_kernel,
                     dim3(nblocks), dim3(256), SMEM_TOTAL, stream,
                     x, weight, bias, out, nrows);
}
